// MultiHeadStructureAttention_6262062318242
// MI455X (gfx1250) — matrix operand order repaired and checked by reading, not compiled, not run on hardware
//
#include <hip/hip_runtime.h>
#include <math.h>

// MI455X / gfx1250, wave32. WMMA f16 16x16x32, f32 accumulate.

typedef __attribute__((ext_vector_type(16))) _Float16 v16h;
typedef __attribute__((ext_vector_type(8)))  float    v8f;

#define NB     4      // batch
#define NH     16     // heads
#define LSEQ   512
#define DMODEL 1024
#define DHEAD  64
#define MROWS  (NB*LSEQ)   // 2048

// ---------------------------------------------------------------------------
// Fragment loaders (wave32; contiguous k order chosen here, not the ISA table order; A and B match)
// A 16x32 f16: lane m=lane&15; hi=lane>>4; halves cover k = hi*16+[0..15]
// (one contiguous 16-half group of a row-major row).
// B 32x16 f16 (stored transposed, row = n): halves cover k = hi*16+[0..15]
// (one contiguous 16-half group).
// C 16x16 f32: elem e -> m = e + hi*8, n = lane&15.
// ---------------------------------------------------------------------------
__device__ __forceinline__ v16h load_a_frag(const _Float16* rowbase, int hi) {
  v16h a;
  const _Float16* p0 = rowbase + hi * 16;
  const _Float16* p1 = rowbase + hi * 16 + 8;
#pragma unroll
  for (int i = 0; i < 8; ++i) { a[i] = p0[i]; a[8 + i] = p1[i]; }
  return a;
}

__device__ __forceinline__ v16h load_b_frag(const _Float16* rowbase, int hi) {
  v16h b;
  const _Float16* p = rowbase + hi * 16;
#pragma unroll
  for (int i = 0; i < 16; ++i) b[i] = p[i];
  return b;
}

// ---------------------------------------------------------------------------
// Kernel 1: convert projection weights f32 -> f16 (done once per launch)
// dst holds wq|wk|wv|wo, each 1024*1024 halfs, contiguous.
// ---------------------------------------------------------------------------
__global__ __launch_bounds__(256) void convert_w_kernel(
    const float* __restrict__ wq, const float* __restrict__ wk,
    const float* __restrict__ wv, const float* __restrict__ wo,
    _Float16* __restrict__ dst) {
  unsigned idx = blockIdx.x * 256u + threadIdx.x;   // 0 .. 4M-1
  unsigned seg = idx >> 20;
  unsigned off = idx & 0xFFFFFu;
  const float* src = (seg == 0) ? wq : (seg == 1) ? wk : (seg == 2) ? wv : wo;
  dst[(size_t)seg * 1048576u + off] = (_Float16)src[off];
}

// ---------------------------------------------------------------------------
// Kernel 2: fused QKV projection GEMM.
// C[2048,1024] = X[2048,1024] @ W[1024,1024] + bias, blockIdx.z selects Q/K/V.
// Block tile 64x128, 8 waves (2x4), each wave 32x32 = 2x2 WMMA fragments.
// Output: head-major f16 [B,H,L,64]; Q scaled by 1/sqrt(dk)=0.125.
// ---------------------------------------------------------------------------
__global__ __launch_bounds__(256) void qkv_proj_kernel(
    const float* __restrict__ xq, const float* __restrict__ xk,
    const float* __restrict__ xv,
    const _Float16* __restrict__ wq_h, const _Float16* __restrict__ wk_h,
    const _Float16* __restrict__ wv_h,
    const float* __restrict__ bq, const float* __restrict__ bk,
    const float* __restrict__ bv,
    _Float16* __restrict__ qh, _Float16* __restrict__ kh,
    _Float16* __restrict__ vh) {
  const int z = blockIdx.z;
  const float*    X    = (z == 0) ? xq : (z == 1) ? xk : xv;
  const _Float16* W    = (z == 0) ? wq_h : (z == 1) ? wk_h : wv_h;
  const float*    bias = (z == 0) ? bq : (z == 1) ? bk : bv;
  _Float16*       out  = (z == 0) ? qh : (z == 1) ? kh : vh;
  const float scale = (z == 0) ? 0.125f : 1.0f;

  __shared__ _Float16 Ah[64 * 40];    // A tile 64x32, stride 40 (pad)
  __shared__ _Float16 Bt[128 * 40];   // B tile transposed: [n][k], stride 40

  const int tid = threadIdx.x;
  const int m0 = blockIdx.y * 64;
  const int n0 = blockIdx.x * 128;
  const int wid = tid >> 5, lane = tid & 31;
  const int l16 = lane & 15, hi = lane >> 4;
  const int wm = wid >> 2, wn = wid & 3;

  v8f c[2][2] = {};

  for (int k0 = 0; k0 < DMODEL; k0 += 32) {
    __syncthreads();
    // stage A (f32 -> f16), 64x32 elements, 8 per thread
#pragma unroll
    for (int i = 0; i < 8; ++i) {
      int e = tid + i * 256;
      int r = e >> 5, cc = e & 31;
      Ah[r * 40 + cc] = (_Float16)X[(size_t)(m0 + r) * DMODEL + k0 + cc];
    }
    // stage W transposed: Bt[n][k] = W[k0+k][n0+n], 32x128, 16 per thread
#pragma unroll
    for (int i = 0; i < 16; ++i) {
      int e = tid + i * 256;
      int kk = e >> 7, n = e & 127;
      Bt[n * 40 + kk] = W[(size_t)(k0 + kk) * DMODEL + n0 + n];
    }
    __syncthreads();

    v16h a[2], bf[2];
#pragma unroll
    for (int mf = 0; mf < 2; ++mf)
      a[mf] = load_a_frag(&Ah[(wm * 32 + mf * 16 + l16) * 40], hi);
#pragma unroll
    for (int nf = 0; nf < 2; ++nf)
      bf[nf] = load_b_frag(&Bt[(wn * 32 + nf * 16 + l16) * 40], hi);
#pragma unroll
    for (int mf = 0; mf < 2; ++mf)
#pragma unroll
      for (int nf = 0; nf < 2; ++nf)
        c[mf][nf] = __builtin_amdgcn_wmma_f32_16x16x32_f16(
            false, a[mf], false, bf[nf], (short)0, c[mf][nf], false, false);
  }

  // epilogue: + bias, * scale, scatter to [B,H,L,64] f16
#pragma unroll
  for (int mf = 0; mf < 2; ++mf)
#pragma unroll
    for (int nf = 0; nf < 2; ++nf)
#pragma unroll
      for (int e = 0; e < 8; ++e) {
        int gm = m0 + wm * 32 + mf * 16 + e + hi * 8;   // row in [0,2048)
        int gn = n0 + wn * 32 + nf * 16 + l16;          // col in [0,1024)
        float val = (c[mf][nf][e] + bias[gn]) * scale;
        int b = gm >> 9, l = gm & 511;
        int h = gn >> 6, d = gn & 63;
        out[(((size_t)(b * NH + h) * LSEQ) + l) * DHEAD + d] = (_Float16)val;
      }
}

// ---------------------------------------------------------------------------
// Kernel 3: flash attention. Block = (b, h, 128 q-rows); 8 waves x 16 q-rows.
// Per 32-key chunk: S = Q Kt (2 chained WMMAs) + bias + graph_bias, online
// softmax (16-lane row reductions), P -> LDS -> A-fragment, PV (4 WMMAs).
// ctx written as f16 [B, L, H*64].
// ---------------------------------------------------------------------------
__global__ __launch_bounds__(256) void attention_kernel(
    const _Float16* __restrict__ qh, const _Float16* __restrict__ kh,
    const _Float16* __restrict__ vh,
    const float* __restrict__ bias, const float* __restrict__ gbias,
    _Float16* __restrict__ ctx) {
  __shared__ _Float16 Kt[32 * 68];       // 32 keys x 64 d, stride 68
  __shared__ _Float16 Vt[32 * 68];
  __shared__ _Float16 Pl[8 * 16 * 40];   // per-wave P tile 16x32, stride 40

  const int tid = threadIdx.x;
  const int wid = tid >> 5, lane = tid & 31;
  const int l16 = lane & 15, hi = lane >> 4;

  const int bh = blockIdx.x >> 2;        // 0..63
  const int qc = blockIdx.x & 3;         // 128-row chunk
  const int b = bh >> 4, h = bh & 15;

  const int qrow0 = qc * 128 + wid * 16;

  const _Float16* qbase = qh + (size_t)(b * NH + h) * LSEQ * DHEAD;
  const _Float16* kbase = kh + (size_t)(b * NH + h) * LSEQ * DHEAD;
  const _Float16* vbase = vh + (size_t)(b * NH + h) * LSEQ * DHEAD;
  const float* biasb = bias  + (size_t)(b * NH + h) * LSEQ * LSEQ;
  const float* gbb   = gbias + (size_t)(b * NH + h) * LSEQ * LSEQ;

  _Float16* Pw = Pl + wid * (16 * 40);

  // Q fragments (A layout), direct from global (row-major, d contiguous)
  const _Float16* qp = qbase + (size_t)(qrow0 + l16) * DHEAD;
  v16h aq0 = load_a_frag(qp, hi);        // d 0..31
  v16h aq1 = load_a_frag(qp + 32, hi);   // d 32..63

  v8f acc[4] = {};
  float mrow[8], lrow[8];
#pragma unroll
  for (int e = 0; e < 8; ++e) { mrow[e] = -1e30f; lrow[e] = 0.0f; }

  for (int jt = 0; jt < LSEQ / 32; ++jt) {
    __syncthreads();
    // cooperative K/V staging: 32 rows x 64 halfs each, 8 halfs/thread
    {
      int row = tid >> 3;
      int cg = (tid & 7) * 8;
      const _Float16* ks = kbase + (size_t)(jt * 32 + row) * DHEAD + cg;
      const _Float16* vs = vbase + (size_t)(jt * 32 + row) * DHEAD + cg;
      _Float16* kd = Kt + row * 68 + cg;
      _Float16* vd = Vt + row * 68 + cg;
#pragma unroll
      for (int i = 0; i < 8; ++i) { kd[i] = ks[i]; vd[i] = vs[i]; }
    }
    __syncthreads();

    // scores for two 16-key sub-tiles
    v8f s[2];
#pragma unroll
    for (int j2 = 0; j2 < 2; ++j2) {
      const _Float16* kr = Kt + (j2 * 16 + l16) * 68;  // B row = key
      v16h b0 = load_b_frag(kr, hi);        // d 0..31
      v16h b1 = load_b_frag(kr + 32, hi);   // d 32..63
      v8f zc = {};
      zc = __builtin_amdgcn_wmma_f32_16x16x32_f16(false, aq0, false, b0,
                                                  (short)0, zc, false, false);
      zc = __builtin_amdgcn_wmma_f32_16x16x32_f16(false, aq1, false, b1,
                                                  (short)0, zc, false, false);
      s[j2] = zc;
    }

    // + mask bias + structural bias (C layout: m = e+hi*8, n = l16)
#pragma unroll
    for (int e = 0; e < 8; ++e) {
      int qr = qrow0 + e + hi * 8;
      int kc0 = jt * 32 + l16;
      int kc1 = kc0 + 16;
      s[0][e] += biasb[(size_t)qr * LSEQ + kc0] + gbb[(size_t)qr * LSEQ + kc0];
      s[1][e] += biasb[(size_t)qr * LSEQ + kc1] + gbb[(size_t)qr * LSEQ + kc1];
    }

    // online softmax over the 32-key chunk
#pragma unroll
    for (int e = 0; e < 8; ++e) {
      float t = fmaxf(s[0][e], s[1][e]);
#pragma unroll
      for (int off = 8; off >= 1; off >>= 1)
        t = fmaxf(t, __shfl_xor(t, off, 16));
      float nm = fmaxf(mrow[e], t);
      float sc = __expf(mrow[e] - nm);
      float p0 = __expf(s[0][e] - nm);
      float p1 = __expf(s[1][e] - nm);
      float rs = p0 + p1;
#pragma unroll
      for (int off = 8; off >= 1; off >>= 1)
        rs += __shfl_xor(rs, off, 16);
      lrow[e] = lrow[e] * sc + rs;
      mrow[e] = nm;
#pragma unroll
      for (int nf = 0; nf < 4; ++nf) acc[nf][e] *= sc;
      // P (C layout) -> LDS row-major 16x32
      int r = e + hi * 8;
      _Float16* pr = Pw + r * 40;
      pr[l16] = (_Float16)p0;
      pr[16 + l16] = (_Float16)p1;
    }

    // intra-wave LDS ordering: P stores must land before A-fragment reads
    __builtin_amdgcn_wave_barrier();
    asm volatile("s_wait_dscnt 0" ::: "memory");

    // PV: A = P(16x32), B = V(32x16 per d-block)
    v16h ap = load_a_frag(Pw + l16 * 40, hi);
#pragma unroll
    for (int nf = 0; nf < 4; ++nf) {
      v16h bvf;
      const _Float16* vb = Vt + hi * 16 * 68 + nf * 16 + l16;
#pragma unroll
      for (int i = 0; i < 16; ++i) bvf[i] = vb[i * 68];
      acc[nf] = __builtin_amdgcn_wmma_f32_16x16x32_f16(
          false, ap, false, bvf, (short)0, acc[nf], false, false);
    }
  }

  // finalize: divide by row sums, write ctx f16 [B, L, H*64]
  float inv[8];
#pragma unroll
  for (int e = 0; e < 8; ++e) inv[e] = 1.0f / lrow[e];
#pragma unroll
  for (int nf = 0; nf < 4; ++nf)
#pragma unroll
    for (int e = 0; e < 8; ++e) {
      int qr = qrow0 + e + hi * 8;
      int col = h * DHEAD + nf * 16 + l16;
      ctx[((size_t)(b * LSEQ + qr) * DMODEL) + col] =
          (_Float16)(acc[nf][e] * inv[e]);
    }
}

// ---------------------------------------------------------------------------
// Kernel 4: output projection GEMM (f16 A) + bo + residual -> f32 preLN
// ---------------------------------------------------------------------------
__global__ __launch_bounds__(256) void out_proj_kernel(
    const _Float16* __restrict__ ctx, const _Float16* __restrict__ wo_h,
    const float* __restrict__ bo, const float* __restrict__ resid,
    float* __restrict__ preln) {
  __shared__ _Float16 Ah[64 * 40];
  __shared__ _Float16 Bt[128 * 40];

  const int tid = threadIdx.x;
  const int m0 = blockIdx.y * 64;
  const int n0 = blockIdx.x * 128;
  const int wid = tid >> 5, lane = tid & 31;
  const int l16 = lane & 15, hi = lane >> 4;
  const int wm = wid >> 2, wn = wid & 3;

  v8f c[2][2] = {};

  for (int k0 = 0; k0 < DMODEL; k0 += 32) {
    __syncthreads();
#pragma unroll
    for (int i = 0; i < 8; ++i) {
      int e = tid + i * 256;
      int r = e >> 5, cc = e & 31;
      Ah[r * 40 + cc] = ctx[(size_t)(m0 + r) * DMODEL + k0 + cc];
    }
#pragma unroll
    for (int i = 0; i < 16; ++i) {
      int e = tid + i * 256;
      int kk = e >> 7, n = e & 127;
      Bt[n * 40 + kk] = wo_h[(size_t)(k0 + kk) * DMODEL + n0 + n];
    }
    __syncthreads();

    v16h a[2], bf[2];
#pragma unroll
    for (int mf = 0; mf < 2; ++mf)
      a[mf] = load_a_frag(&Ah[(wm * 32 + mf * 16 + l16) * 40], hi);
#pragma unroll
    for (int nf = 0; nf < 2; ++nf)
      bf[nf] = load_b_frag(&Bt[(wn * 32 + nf * 16 + l16) * 40], hi);
#pragma unroll
    for (int mf = 0; mf < 2; ++mf)
#pragma unroll
      for (int nf = 0; nf < 2; ++nf)
        c[mf][nf] = __builtin_amdgcn_wmma_f32_16x16x32_f16(
            false, a[mf], false, bf[nf], (short)0, c[mf][nf], false, false);
  }

#pragma unroll
  for (int mf = 0; mf < 2; ++mf)
#pragma unroll
    for (int nf = 0; nf < 2; ++nf)
#pragma unroll
      for (int e = 0; e < 8; ++e) {
        int gm = m0 + wm * 32 + mf * 16 + e + hi * 8;
        int gn = n0 + wn * 32 + nf * 16 + l16;
        size_t idx = (size_t)gm * DMODEL + gn;
        preln[idx] = c[mf][nf][e] + bo[gn] + resid[idx];
      }
}

// ---------------------------------------------------------------------------
// Kernel 5: LayerNorm over D=1024 per row (2048 rows), eps=1e-6
// ---------------------------------------------------------------------------
__global__ __launch_bounds__(256) void layernorm_kernel(
    const float* __restrict__ preln, const float* __restrict__ g,
    const float* __restrict__ bb, float* __restrict__ out) {
  const int row = blockIdx.x;
  const float* x = preln + (size_t)row * DMODEL;
  float s = 0.0f, s2 = 0.0f;
  for (int i = threadIdx.x; i < DMODEL; i += 256) {
    float t = x[i];
    s += t;
    s2 += t * t;
  }
#pragma unroll
  for (int off = 16; off >= 1; off >>= 1) {
    s  += __shfl_xor(s, off, 32);
    s2 += __shfl_xor(s2, off, 32);
  }
  __shared__ float red[2][8];
  int wid = threadIdx.x >> 5, lane = threadIdx.x & 31;
  if (lane == 0) { red[0][wid] = s; red[1][wid] = s2; }
  __syncthreads();
  s = 0.0f; s2 = 0.0f;
#pragma unroll
  for (int i = 0; i < 8; ++i) { s += red[0][i]; s2 += red[1][i]; }
  float mu = s * (1.0f / DMODEL);
  float var = s2 * (1.0f / DMODEL) - mu * mu;
  float inv = rsqrtf(var + 1e-6f);
  for (int i = threadIdx.x; i < DMODEL; i += 256)
    out[(size_t)row * DMODEL + i] = (x[i] - mu) * inv * g[i] + bb[i];
}

// ---------------------------------------------------------------------------
// Host-side launch
// ---------------------------------------------------------------------------
extern "C" void kernel_launch(void* const* d_in, const int* in_sizes, int n_in,
                              void* d_out, int out_size, void* d_ws,
                              size_t ws_size, hipStream_t stream) {
  const float* q     = (const float*)d_in[0];
  const float* k     = (const float*)d_in[1];
  const float* v     = (const float*)d_in[2];
  const float* bias  = (const float*)d_in[3];
  const float* gbias = (const float*)d_in[4];
  const float* wq    = (const float*)d_in[5];
  const float* bq    = (const float*)d_in[6];
  const float* wk    = (const float*)d_in[7];
  const float* bk    = (const float*)d_in[8];
  const float* wv    = (const float*)d_in[9];
  const float* bv    = (const float*)d_in[10];
  const float* wo    = (const float*)d_in[11];
  const float* bo    = (const float*)d_in[12];
  const float* lng   = (const float*)d_in[13];
  const float* lnb   = (const float*)d_in[14];
  float* out = (float*)d_out;

  // workspace layout (32 MB total)
  _Float16* wq_h = (_Float16*)d_ws;              // 1M halfs
  _Float16* wk_h = wq_h + (1u << 20);
  _Float16* wv_h = wk_h + (1u << 20);
  _Float16* wo_h = wv_h + (1u << 20);
  _Float16* qh   = wo_h + (1u << 20);            // B*H*L*64 = 2,097,152 halfs
  _Float16* kh   = qh + 2097152u;
  _Float16* vh   = kh + 2097152u;
  _Float16* ctx  = vh + 2097152u;                // [B, L, 1024] halfs
  float*    preln = (float*)(ctx + 2097152u);    // 2048*1024 f32

  convert_w_kernel<<<16384, 256, 0, stream>>>(wq, wk, wv, wo, wq_h);
  qkv_proj_kernel<<<dim3(8, 32, 3), 256, 0, stream>>>(
      q, k, v, wq_h, wk_h, wv_h, bq, bk, bv, qh, kh, vh);
  attention_kernel<<<256, 256, 0, stream>>>(qh, kh, vh, bias, gbias, ctx);
  out_proj_kernel<<<dim3(8, 32), 256, 0, stream>>>(ctx, wo_h, bo, q, preln);
  layernorm_kernel<<<2048, 256, 0, stream>>>(preln, lng, lnb, out);
}
